// DeepKernelRegressionModel_85014582657267
// MI455X (gfx1250) — matrix operand order repaired and checked by reading, not compiled, not run on hardware
//
#include <hip/hip_runtime.h>

typedef __attribute__((ext_vector_type(16))) _Float16 v16h;
typedef __attribute__((ext_vector_type(8)))  _Float16 v8h;
typedef __attribute__((ext_vector_type(8)))  float    v8f;

#define D_IN  64
#define H_DIM 32
#define T_OUT 8

// ---------------------------------------------------------------------------
// Kernel 1: 3-layer MLP featurizer. One row per thread; weights in LDS.
// Emits f16 features (for WMMA) + exact fp32 squared row norms.
// ---------------------------------------------------------------------------
__global__ __launch_bounds__(256) void mlp_featurize(
    const float* __restrict__ In,
    const float* __restrict__ W1, const float* __restrict__ b1,
    const float* __restrict__ W2, const float* __restrict__ b2,
    const float* __restrict__ W3, const float* __restrict__ b3,
    _Float16* __restrict__ Fh, float* __restrict__ norms, int nrows)
{
    __shared__ float sW1[D_IN * H_DIM];
    __shared__ float sW2[H_DIM * H_DIM];
    __shared__ float sW3[H_DIM * H_DIM];
    __shared__ float sb1[H_DIM], sb2[H_DIM], sb3[H_DIM];

    const int t = threadIdx.x;
    for (int i = t; i < D_IN * H_DIM; i += 256) sW1[i] = W1[i];
    for (int i = t; i < H_DIM * H_DIM; i += 256) { sW2[i] = W2[i]; sW3[i] = W3[i]; }
    if (t < H_DIM) { sb1[t] = b1[t]; sb2[t] = b2[t]; sb3[t] = b3[t]; }
    __syncthreads();

    const int row = blockIdx.x * 256 + t;
    if (row >= nrows) return;

    float x[D_IN];
    const float* rp = In + (size_t)row * D_IN;
    #pragma unroll
    for (int d = 0; d < D_IN; ++d) x[d] = rp[d];

    float h1[H_DIM];
    for (int h = 0; h < H_DIM; ++h) {
        float acc = sb1[h];
        #pragma unroll
        for (int d = 0; d < D_IN; ++d) acc = fmaf(x[d], sW1[d * H_DIM + h], acc);
        h1[h] = fmaxf(acc, 0.0f);
    }
    float h2[H_DIM];
    for (int h = 0; h < H_DIM; ++h) {
        float acc = sb2[h];
        #pragma unroll
        for (int d = 0; d < H_DIM; ++d) acc = fmaf(h1[d], sW2[d * H_DIM + h], acc);
        h2[h] = fmaxf(acc, 0.0f);
    }
    float nrm = 0.0f;
    _Float16* op = Fh + (size_t)row * H_DIM;
    for (int h = 0; h < H_DIM; ++h) {
        float acc = sb3[h];
        #pragma unroll
        for (int d = 0; d < H_DIM; ++d) acc = fmaf(h2[d], sW3[d * H_DIM + h], acc);
        acc = fmaxf(acc, 0.0f);
        nrm = fmaf(acc, acc, nrm);
        op[h] = (_Float16)acc;
    }
    norms[row] = nrm;
}

// ---------------------------------------------------------------------------
// Kernel 2: fused pairwise RBF kernel + row-normalized regression.
// One wave32 per 16-row tile of Xf; loops over all M in 16-col chunks.
// Per chunk: one v_wmma_f32_16x16x32_f16 gives the 16x16 cross-term tile,
// then sq = ||x||^2 + ||y||^2 - 2 x.y, k = 2^(-sq/(2 ln2)) via v_exp_f32,
// and register accumulation of num[16][8] (k * Y_target) and den[16].
// Never materializes the [N,M] kernel matrix (would be 256 MB > L2).
// ---------------------------------------------------------------------------
__global__ __launch_bounds__(128) void dkr_fused(
    const _Float16* __restrict__ Xf, const float* __restrict__ xn,
    const _Float16* __restrict__ Yf, const float* __restrict__ yn,
    const float* __restrict__ Yt, float* __restrict__ Out,
    int M, int ntiles)
{
    const int lane = threadIdx.x & 31;
    const int wave = threadIdx.x >> 5;
    const int tile = blockIdx.x * (blockDim.x >> 5) + wave;
    if (tile >= ntiles) return;
    const int row0 = tile << 4;
    const int hh = lane >> 4;   // lane half: selects K group / M group
    const int ln = lane & 15;   // A: row index, B: column index

    // A-matrix 16x32 f16 layout: lane = M row; per lane-half h:
    // VGPR0-7 hold K = 16h..16h+15 (contiguous, matching B).
    const _Float16* ap = Xf + (size_t)(row0 + ln) * H_DIM;
    v8h alo = *(const v8h*)(ap + 16 * hh);
    v8h ahi = *(const v8h*)(ap + 16 * hh + 8);
    v16h A;
    #pragma unroll
    for (int i = 0; i < 8; ++i) { A[i] = alo[i]; A[i + 8] = ahi[i]; }

    // C/D layout: VGPR v, lane half h -> M = v + 8h, N = lane % 16.
    float xnv[8];
    #pragma unroll
    for (int v = 0; v < 8; ++v) xnv[v] = xn[row0 + v + 8 * hh];

    float num[8][8];
    float den[8];
    #pragma unroll
    for (int v = 0; v < 8; ++v) {
        den[v] = 0.0f;
        #pragma unroll
        for (int tt = 0; tt < T_OUT; ++tt) num[v][tt] = 0.0f;
    }

    const float c_exp = -0.72134752044448170f;  // -log2(e)/2 : exp(-sq/2)

    for (int j = 0; j < M; j += 16) {
        // B-matrix 32x16 f16: lane = N column; lanes 0-15 hold K=0..15,
        // lanes 16-31 hold K=16..31 (contiguous per half).
        const _Float16* bp = Yf + (size_t)(j + ln) * H_DIM + 16 * hh;
        v8h blo = *(const v8h*)(bp);
        v8h bhi = *(const v8h*)(bp + 8);
        v16h B;
        #pragma unroll
        for (int i = 0; i < 8; ++i) { B[i] = blo[i]; B[i + 8] = bhi[i]; }

        v8f C = {};
        C = __builtin_amdgcn_wmma_f32_16x16x32_f16(
                /*neg_a=*/false, A, /*neg_b=*/false, B,
                /*c_mod=*/(short)0, C, /*reuse_a=*/false, /*reuse_b=*/false);

        const float ynv = yn[j + ln];
        const float4 yt0 = *(const float4*)(Yt + (size_t)(j + ln) * T_OUT);
        const float4 yt1 = *(const float4*)(Yt + (size_t)(j + ln) * T_OUT + 4);
        const float yt[8] = { yt0.x, yt0.y, yt0.z, yt0.w,
                              yt1.x, yt1.y, yt1.z, yt1.w };

        #pragma unroll
        for (int v = 0; v < 8; ++v) {
            float sq = xnv[v] + ynv - 2.0f * C[v];
            sq = fmaxf(sq, 0.0f);
            const float k = __builtin_amdgcn_exp2f(sq * c_exp);
            den[v] += k;
            #pragma unroll
            for (int tt = 0; tt < T_OUT; ++tt)
                num[v][tt] = fmaf(k, yt[tt], num[v][tt]);
        }
    }

    // Reduce over the 16 N-columns held by each 16-lane half. XOR masks
    // 1,2,4,8 keep bit 4 (the M-half select) intact under wave32.
    #pragma unroll
    for (int m = 1; m <= 8; m <<= 1) {
        #pragma unroll
        for (int v = 0; v < 8; ++v) {
            den[v] += __shfl_xor(den[v], m, 32);
            #pragma unroll
            for (int tt = 0; tt < T_OUT; ++tt)
                num[v][tt] += __shfl_xor(num[v][tt], m, 32);
        }
    }

    if (ln == 0) {  // lane 0 -> rows row0..row0+7, lane 16 -> rows +8..+15
        #pragma unroll
        for (int v = 0; v < 8; ++v) {
            const int row = row0 + v + 8 * hh;
            const float inv = 1.0f / den[v];
            #pragma unroll
            for (int tt = 0; tt < T_OUT; ++tt)
                Out[(size_t)row * T_OUT + tt] = num[v][tt] * inv;
        }
    }
}

// ---------------------------------------------------------------------------
extern "C" void kernel_launch(void* const* d_in, const int* in_sizes, int n_in,
                              void* d_out, int out_size, void* d_ws, size_t ws_size,
                              hipStream_t stream) {
    const float* X  = (const float*)d_in[0];
    const float* Y  = (const float*)d_in[1];
    const float* Yt = (const float*)d_in[2];
    const float* W1 = (const float*)d_in[3];
    const float* b1 = (const float*)d_in[4];
    const float* W2 = (const float*)d_in[5];
    const float* b2 = (const float*)d_in[6];
    const float* W3 = (const float*)d_in[7];
    const float* b3 = (const float*)d_in[8];
    float* Out = (float*)d_out;

    const int N = in_sizes[0] / D_IN;
    const int M = in_sizes[1] / D_IN;

    char* ws = (char*)d_ws;
    size_t off = 0;
    _Float16* Xf = (_Float16*)(ws + off); off += (size_t)N * H_DIM * sizeof(_Float16);
    _Float16* Yf = (_Float16*)(ws + off); off += (size_t)M * H_DIM * sizeof(_Float16);
    float*    xn = (float*)(ws + off);    off += (size_t)N * sizeof(float);
    float*    yn = (float*)(ws + off);    off += (size_t)M * sizeof(float);

    mlp_featurize<<<(N + 255) / 256, 256, 0, stream>>>(X, W1, b1, W2, b2, W3, b3, Xf, xn, N);
    mlp_featurize<<<(M + 255) / 256, 256, 0, stream>>>(Y, W1, b1, W2, b2, W3, b3, Yf, yn, M);

    const int ntiles = N / 16;                 // 512 wave-tiles
    const int wavesPerBlock = 4;               // 128 threads
    const int blocks = (ntiles + wavesPerBlock - 1) / wavesPerBlock;
    dkr_fused<<<blocks, wavesPerBlock * 32, 0, stream>>>(Xf, xn, Yf, yn, Yt, Out, M, ntiles);
}
